// KimiDeltaAttention_3246995276383
// MI455X (gfx1250) — matrix operand order repaired and checked by reading, not compiled, not run on hardware
//
#include <hip/hip_runtime.h>
#include <math.h>

// ---------------------------------------------------------------------------
// Problem constants (match reference)
// ---------------------------------------------------------------------------
static constexpr int Bb   = 2;
static constexpr int Tt   = 2048;
static constexpr int HID  = 2048;
static constexpr int Hh   = 16;
static constexpr int DK   = 128;
static constexpr int DV   = 128;
static constexpr int GLR  = 128;
static constexpr int KD   = Hh * DK;   // 2048
static constexpr int VD   = Hh * DV;   // 2048
static constexpr int DCONV = 4;
static constexpr int BT   = Bb * Tt;   // 4096 rows
static constexpr float EPS = 1e-6f;

typedef __attribute__((ext_vector_type(16))) _Float16 v16h;
typedef __attribute__((ext_vector_type(4)))  _Float16 h4;
typedef __attribute__((ext_vector_type(2)))  _Float16 h2;
typedef __attribute__((ext_vector_type(8)))  float    v8f;

// ---------------------------------------------------------------------------
// WMMA GEMM:  C[M,N] = A[M,K] * B[K,N]  (fp32 in/out, f16 WMMA compute, f32
// accumulate).  Block = 128 threads = 4 waves; tile 64x64, K-step 32,
// double-buffered LDS staging with register prefetch of the next tile so
// global loads overlap WMMA issue (one barrier per K-step).
//
// Tiles are staged into LDS in *fragment order*, so each lane reads its whole
// 16-half WMMA operand as one contiguous 32-byte load (2x ds_load_b128).
//
// Fragment layouts used here (CDNA5 16-bit, 16x16x32; contiguous k, not ISA table order):
//   A (16x32): lane = half*16 + m ; element i -> K = half*16 + i.
//              Inverse: half=K>>4,
//              i=K&15  -> float4 K-group = contiguous i.
//   B (32x16): lane = half*16 + n ; element i -> K = half*16 + i.
//   C/D      : VGPR r -> row (r + half*8), col = lane&15.
//
// Requires M % 64 == 0, K % 32 == 0, N % 4 == 0 (true for all launches here).
// ---------------------------------------------------------------------------
__device__ __forceinline__ void kda_load_tile(
    const float* __restrict__ A, const float* __restrict__ Bm,
    int N, int K, int rowBase, int colBase, int kt, int tid,
    float4 (&ar)[4], float4 (&br)[2][2]) {
    #pragma unroll
    for (int g = 0; g < 4; ++g) {
        int e = (g * 128 + tid) * 4;       // element index in 64x32 A tile
        int r = e >> 5, c = e & 31;
        ar[g] = *reinterpret_cast<const float4*>(
            A + (size_t)(rowBase + r) * K + kt + c);
    }
    #pragma unroll
    for (int g = 0; g < 2; ++g) {
        int ch = g * 128 + tid;            // chunk = 2 rows x 4 cols of B tile
        int r0 = (ch >> 4) * 2;
        int c4 = (ch & 15) * 4;
        int gc = colBase + c4;
        float4 z = (float4){0.f, 0.f, 0.f, 0.f};
        br[g][0] = z;
        br[g][1] = z;
        if (gc < N) {                      // N % 4 == 0: group fully in/out
            br[g][0] = *reinterpret_cast<const float4*>(Bm + (size_t)(kt + r0    ) * N + gc);
            br[g][1] = *reinterpret_cast<const float4*>(Bm + (size_t)(kt + r0 + 1) * N + gc);
        }
    }
}

__device__ __forceinline__ void kda_store_tile(
    _Float16* Af, _Float16* Bf, int tid,
    const float4 (&ar)[4], const float4 (&br)[2][2]) {
    #pragma unroll
    for (int g = 0; g < 4; ++g) {          // A: one ds_store_b64 per group
        int e = (g * 128 + tid) * 4;
        int r = e >> 5, c = e & 31;
        int w    = r >> 4;
        int m    = r & 15;
        int half = c >> 4;
        int ib   = c & 15;
        _Float16* dst = Af + ((w * 32 + half * 16 + m) << 4) + ib;
        *reinterpret_cast<h4*>(dst) =
            (h4){(_Float16)ar[g].x, (_Float16)ar[g].y,
                 (_Float16)ar[g].z, (_Float16)ar[g].w};
    }
    #pragma unroll
    for (int g = 0; g < 2; ++g) {          // B: packed K-pair ds_store_b32
        int ch = g * 128 + tid;
        int r0 = (ch >> 4) * 2;
        int c4 = (ch & 15) * 4;
        int half = r0 >> 4;
        int i0   = r0 & 15;                // even
        #pragma unroll
        for (int u = 0; u < 4; ++u) {
            int cc     = c4 + u;
            int lane_b = half * 16 + (cc & 15);
            int j      = cc >> 4;
            _Float16* dst = Bf + ((j * 32 + lane_b) << 4) + i0;
            float e0 = (u == 0) ? br[g][0].x : (u == 1) ? br[g][0].y
                     : (u == 2) ? br[g][0].z : br[g][0].w;
            float e1 = (u == 0) ? br[g][1].x : (u == 1) ? br[g][1].y
                     : (u == 2) ? br[g][1].z : br[g][1].w;
            *reinterpret_cast<h2*>(dst) = (h2){(_Float16)e0, (_Float16)e1};
        }
    }
}

__global__ __launch_bounds__(128)
void kda_gemm_f16wmma(const float* __restrict__ A,
                      const float* __restrict__ Bm,
                      float* __restrict__ C,
                      int M, int N, int K) {
    __shared__ v16h AfragV[2 * 128];   // double-buffered: 2 x (4 tiles x 32 lanes)
    __shared__ v16h BfragV[2 * 128];

    const int tid  = threadIdx.x;
    const int wave = tid >> 5;
    const int lane = tid & 31;

    const int rowBase = blockIdx.y * 64;
    const int colBase = blockIdx.x * 64;

    v8f acc[4];
    #pragma unroll
    for (int j = 0; j < 4; ++j)
        acc[j] = (v8f){0.f,0.f,0.f,0.f,0.f,0.f,0.f,0.f};

    float4 curA[4], nxtA[4];
    float4 curB[2][2], nxtB[2][2];
    #pragma unroll
    for (int g = 0; g < 4; ++g) nxtA[g] = (float4){0.f,0.f,0.f,0.f};
    #pragma unroll
    for (int g = 0; g < 2; ++g)
        nxtB[g][0] = nxtB[g][1] = (float4){0.f,0.f,0.f,0.f};

    kda_load_tile(A, Bm, N, K, rowBase, colBase, 0, tid, curA, curB);

    int p = 0;
    for (int kt = 0; kt < K; kt += 32) {
        // stage current tile (registers -> fragment-order LDS, buffer p)
        kda_store_tile(reinterpret_cast<_Float16*>(AfragV + p * 128),
                       reinterpret_cast<_Float16*>(BfragV + p * 128),
                       tid, curA, curB);
        __syncthreads();

        // issue next tile's global loads (waits land before next iteration's
        // converts, overlapping HBM latency with the WMMAs below)
        const bool more = (kt + 32) < K;
        if (more)
            kda_load_tile(A, Bm, N, K, rowBase, colBase, kt + 32, tid, nxtA, nxtB);

        // distance-2 prefetch (emits global_prefetch_b8)
        if (kt + 64 < K) {
            __builtin_prefetch(A + (size_t)(rowBase + (tid >> 1)) * K
                                 + kt + 64 + (tid & 1) * 16, 0, 3);
            int pc = colBase + (tid & 3) * 16;
            if (pc < N)
                __builtin_prefetch(Bm + (size_t)(kt + 64 + (tid >> 2)) * N + pc, 0, 3);
        }

        // fragments: contiguous 32B per lane (2x ds_load_b128) -> WMMA
        const v16h afrag = AfragV[p * 128 + wave * 32 + lane];
        #pragma unroll
        for (int j = 0; j < 4; ++j) {
            const v16h bfrag = BfragV[p * 128 + j * 32 + lane];
            acc[j] = __builtin_amdgcn_wmma_f32_16x16x32_f16(
                false, afrag, false, bfrag, (short)0, acc[j], false, false);
        }

        #pragma unroll
        for (int g = 0; g < 4; ++g) curA[g] = nxtA[g];
        #pragma unroll
        for (int g = 0; g < 2; ++g) {
            curB[g][0] = nxtB[g][0];
            curB[g][1] = nxtB[g][1];
        }
        p ^= 1;
    }

    // ---- writeback ---------------------------------------------------------
    const int half = (lane >> 4) & 1;
    const int l16  = lane & 15;
    #pragma unroll
    for (int j = 0; j < 4; ++j) {
        #pragma unroll
        for (int r = 0; r < 8; ++r) {
            int row = rowBase + wave * 16 + r + half * 8;
            int col = colBase + j * 16 + l16;
            if (col < N)
                C[(size_t)row * N + col] = acc[j][r];
        }
    }
}

// ---------------------------------------------------------------------------
// Depthwise causal conv (DCONV=4) + SiLU + optional per-head L2 norm.
// grid = (B*T, C/128), block = 128.  One block handles one (b,t,head).
// ---------------------------------------------------------------------------
__global__ __launch_bounds__(128)
void kda_conv_silu_norm(const float* __restrict__ proj,
                        const float* __restrict__ w,     // [DCONV, C]
                        float* __restrict__ out,
                        int C, int do_norm, float scale) {
    const int bt = blockIdx.x;
    const int h  = blockIdx.y;
    const int d  = threadIdx.x;
    const int c  = h * 128 + d;
    const int b  = bt / Tt;
    const int t  = bt % Tt;

    float s = 0.f;
    #pragma unroll
    for (int j = 0; j < DCONV; ++j) {
        int tt = t - (DCONV - 1) + j;
        if (tt >= 0)
            s += w[j * C + c] * proj[((size_t)b * Tt + tt) * C + c];
    }
    // SiLU
    s = s / (1.f + __expf(-s));

    if (do_norm) {
        __shared__ float red[4];
        float ss = s * s;
        #pragma unroll
        for (int off = 16; off >= 1; off >>= 1)
            ss += __shfl_xor(ss, off, 32);
        if ((threadIdx.x & 31) == 0) red[threadIdx.x >> 5] = ss;
        __syncthreads();
        float tot = red[0] + red[1] + red[2] + red[3];
        s = s * rsqrtf(tot + EPS) * scale;
    }
    out[(size_t)bt * C + c] = s;
}

// ---------------------------------------------------------------------------
// Decay gate + beta:  g = -exp(A_log[h]) * softplus(glin + dt_bias[h]),
//                     beta = sigmoid(blin).   n = B*T*H elements.
// ---------------------------------------------------------------------------
__global__ __launch_bounds__(256)
void kda_gates(const float* __restrict__ glin,
               const float* __restrict__ blin,
               const float* __restrict__ A_log,
               const float* __restrict__ dt_bias,
               float* __restrict__ g,
               float* __restrict__ beta,
               int n) {
    int i = blockIdx.x * blockDim.x + threadIdx.x;
    if (i >= n) return;
    int h = i % Hh;
    float x  = glin[i] + dt_bias[h];
    float sp = (x > 20.f) ? x : log1pf(__expf(x));
    g[i]    = -__expf(A_log[h]) * sp;
    beta[i] = 1.f / (1.f + __expf(-blin[i]));
}

// ---------------------------------------------------------------------------
// Gated delta-rule recurrence.  One workgroup per (b,h); state S[128][128]
// lives in LDS (64 KB, fine under CDNA5's 320 KB/WGP).  256 threads:
// thread = (half, v) with half = tid>>7, v = tid&127; each sweeps 64 k-rows
// of column v.  Two LDS sweeps per step: (decay + k^T S) and (update + q^T S).
// ---------------------------------------------------------------------------
__global__ __launch_bounds__(256)
void kda_recurrence(const float* __restrict__ qbuf,   // [B,T,H,DK]
                    const float* __restrict__ kbuf,
                    const float* __restrict__ vbuf,
                    const float* __restrict__ gbuf,   // [B,T,H]
                    const float* __restrict__ betabuf,
                    float* __restrict__ obuf) {       // [B,T,H,DV]
    extern __shared__ float lds[];
    float* S    = lds;                 // 128*128
    float* part = S + 128 * 128;       // 256
    float* sK   = part + 256;          // 128
    float* sQ   = sK + 128;            // 128
    float* sD   = sQ + 128;            // 128

    const int bh = blockIdx.x;
    const int b  = bh / Hh;
    const int h  = bh % Hh;
    const int tid  = threadIdx.x;
    const int v    = tid & 127;
    const int half = tid >> 7;
    const int kbase = half * 64;

    for (int i = tid; i < 128 * 128; i += 256) S[i] = 0.f;
    __syncthreads();

    for (int t = 0; t < Tt; ++t) {
        const size_t rowoff = (((size_t)b * Tt + t) * Hh + h) * 128;
        if (tid < 128) { sK[tid] = kbuf[rowoff + tid]; sQ[tid] = qbuf[rowoff + tid]; }
        const float eg = __expf(gbuf[((size_t)b * Tt + t) * Hh + h]);
        const float bv = betabuf[((size_t)b * Tt + t) * Hh + h];
        __syncthreads();

        // sweep 1: S *= exp(g);  kv[v] = sum_k k[k] * S[k][v]
        float acc = 0.f;
        #pragma unroll 4
        for (int kk = 0; kk < 64; ++kk) {
            int k = kbase + kk;
            float s = S[k * 128 + v] * eg;
            S[k * 128 + v] = s;
            acc += sK[k] * s;
        }
        part[tid] = acc;
        __syncthreads();

        if (half == 0) {
            float kv = part[v] + part[128 + v];
            sD[v] = (vbuf[rowoff + v] - kv) * bv;
        }
        __syncthreads();

        // sweep 2: S += k outer delta;  o[v] = sum_k q[k] * S[k][v]
        float acc2 = 0.f;
        #pragma unroll 4
        for (int kk = 0; kk < 64; ++kk) {
            int k = kbase + kk;
            float s = S[k * 128 + v] + sK[k] * sD[v];
            S[k * 128 + v] = s;
            acc2 += sQ[k] * s;
        }
        part[tid] = acc2;
        __syncthreads();

        if (half == 0) obuf[rowoff + v] = part[v] + part[128 + v];
        __syncthreads();
    }
}

// ---------------------------------------------------------------------------
// Gated RMSNorm (per head over DV) * silu(gate).  In-place on o.
// grid = (B*T, H), block = 128.
// ---------------------------------------------------------------------------
__global__ __launch_bounds__(128)
void kda_rmsnorm_gate(float* __restrict__ o,
                      const float* __restrict__ gate,   // [B*T, VD]
                      const float* __restrict__ norm_kernel) { // [DV]
    const int bt = blockIdx.x;
    const int h  = blockIdx.y;
    const int d  = threadIdx.x;
    const size_t idx = ((size_t)bt * Hh + h) * 128 + d;

    float ov = o[idx];
    __shared__ float red[4];
    float ss = ov * ov;
    #pragma unroll
    for (int off = 16; off >= 1; off >>= 1)
        ss += __shfl_xor(ss, off, 32);
    if ((threadIdx.x & 31) == 0) red[threadIdx.x >> 5] = ss;
    __syncthreads();
    float var = (red[0] + red[1] + red[2] + red[3]) * (1.f / 128.f);

    float gt = gate[(size_t)bt * VD + h * 128 + d];
    float silu_g = gt / (1.f + __expf(-gt));
    o[idx] = ov * rsqrtf(var + EPS) * norm_kernel[d] * silu_g;
}

// ---------------------------------------------------------------------------
// Host-side launch
// ---------------------------------------------------------------------------
static inline void launch_gemm(const float* A, const float* B, float* C,
                               int M, int N, int K, hipStream_t s) {
    dim3 grid((N + 63) / 64, (M + 63) / 64);
    kda_gemm_f16wmma<<<grid, 128, 0, s>>>(A, B, C, M, N, K);
}

extern "C" void kernel_launch(void* const* d_in, const int* in_sizes, int n_in,
                              void* d_out, int out_size, void* d_ws, size_t ws_size,
                              hipStream_t stream) {
    // inputs, in setup_inputs() order
    const float* x        = (const float*)d_in[0];   // [B,T,HID]
    const float* Wq       = (const float*)d_in[1];   // [HID,KD]
    const float* Wk       = (const float*)d_in[2];
    const float* Wv       = (const float*)d_in[3];
    const float* conv_q   = (const float*)d_in[4];   // [DCONV,KD]
    const float* conv_k   = (const float*)d_in[5];
    const float* conv_v   = (const float*)d_in[6];
    const float* Wfa      = (const float*)d_in[7];   // [HID,GLR]
    const float* Wfb      = (const float*)d_in[8];   // [GLR,H]
    const float* Wb       = (const float*)d_in[9];   // [HID,H]
    const float* Wga      = (const float*)d_in[10];  // [HID,GLR]
    const float* Wgb      = (const float*)d_in[11];  // [GLR,VD]
    const float* Wo       = (const float*)d_in[12];  // [VD,HID]
    const float* A_log    = (const float*)d_in[13];  // [H]
    const float* dt_bias  = (const float*)d_in[14];  // [H]
    const float* norm_k   = (const float*)d_in[15];  // [DV]

    float* out = (float*)d_out;                      // [B,T,HID] fp32

    // workspace layout (floats)
    float* ws = (float*)d_ws;
    size_t off = 0;
    const size_t big = (size_t)BT * KD;              // 4096*2048
    float* qp   = ws + off; off += big;              // raw q projection
    float* kp   = ws + off; off += big;
    float* vp   = ws + off; off += big;
    float* qn   = ws + off; off += big;              // post conv+silu+norm
    float* kn   = ws + off; off += big;
    float* vn   = ws + off; off += big;
    float* gate = ws + off; off += big;              // xga @ Wgb
    float* obuf = ws + off; off += big;              // recurrence output
    float* xfa  = ws + off; off += (size_t)BT * GLR;
    float* xga  = ws + off; off += (size_t)BT * GLR;
    float* glin = ws + off; off += (size_t)BT * Hh;
    float* blin = ws + off; off += (size_t)BT * Hh;
    float* gdec = ws + off; off += (size_t)BT * Hh;
    float* beta = ws + off; off += (size_t)BT * Hh;

    // --- 1. projections (WMMA GEMMs) ---------------------------------------
    launch_gemm(x, Wq, qp, BT, KD, HID, stream);
    launch_gemm(x, Wk, kp, BT, KD, HID, stream);
    launch_gemm(x, Wv, vp, BT, VD, HID, stream);
    launch_gemm(x, Wfa, xfa, BT, GLR, HID, stream);
    launch_gemm(x, Wga, xga, BT, GLR, HID, stream);
    launch_gemm(x, Wb, blin, BT, Hh, HID, stream);
    launch_gemm(xfa, Wfb, glin, BT, Hh, GLR, stream);
    launch_gemm(xga, Wgb, gate, BT, VD, GLR, stream);

    // --- 2. conv + silu (+ l2 norm for q,k) --------------------------------
    {
        dim3 grid(BT, KD / 128);
        const float qscale = rsqrtf((float)DK);  // DK^-0.5 applied after l2norm
        kda_conv_silu_norm<<<grid, 128, 0, stream>>>(qp, conv_q, qn, KD, 1, qscale);
        kda_conv_silu_norm<<<grid, 128, 0, stream>>>(kp, conv_k, kn, KD, 1, 1.0f);
        kda_conv_silu_norm<<<grid, 128, 0, stream>>>(vp, conv_v, vn, VD, 0, 1.0f);
    }

    // --- 3. decay gate + beta ----------------------------------------------
    {
        int n = BT * Hh;
        kda_gates<<<(n + 255) / 256, 256, 0, stream>>>(glin, blin, A_log, dt_bias,
                                                       gdec, beta, n);
    }

    // --- 4. sequential gated delta-rule scan (LDS-resident state) ----------
    {
        size_t ldsBytes = (size_t)(128 * 128 + 256 + 3 * 128) * sizeof(float); // ~68 KB
        hipFuncSetAttribute((const void*)kda_recurrence,
                            hipFuncAttributeMaxDynamicSharedMemorySize,
                            (int)ldsBytes);
        kda_recurrence<<<Bb * Hh, 256, ldsBytes, stream>>>(qn, kn, vn, gdec, beta, obuf);
    }

    // --- 5. gated RMSNorm (in place on obuf) -------------------------------
    {
        dim3 grid(BT, Hh);
        kda_rmsnorm_gate<<<grid, 128, 0, stream>>>(obuf, gate, norm_k);
    }

    // --- 6. output projection ----------------------------------------------
    launch_gemm(obuf, Wo, out, BT, HID, VD, stream);
}
